// UnifiedLSTM_36910948942108
// MI455X (gfx1250) — matrix operand order repaired and checked by reading, not compiled, not run on hardware
//
#include <hip/hip_runtime.h>

// ---------------------------------------------------------------------------
// Bidirectional LSTM for MI455X (gfx1250, wave32, WMMA + async-to-LDS).
//   1. Convert x / transpose-convert weights to bf16 (W^T [N][K] so B columns
//      are contiguous).
//   2. xp = x @ Wx + bx (both dirs): parallel bf16 WMMA GEMM, B panel staged
//      into LDS with global_load_async_to_lds_b128 (ASYNCcnt + barrier).
//   3. 256 step kernels (kernel boundary = grid sync). Block = one (dir,
//      unit-tile); its 8 waves cover all batch tiles and share the Wh panel
//      (single L2 read per panel per step). LSTM nonlinearity fused in the
//      WMMA epilogue. h double-buffered bf16, c stays f32.
//   4. merged hs [S][B][2H] bf16 @ Wfc via the same LDS-staged WMMA GEMM.
//   5. memcpyAsync final h/c states into the output tail.
// ---------------------------------------------------------------------------

typedef __bf16 bf16_t;
typedef __attribute__((ext_vector_type(16))) __bf16 v16bf;
typedef __attribute__((ext_vector_type(8)))  __bf16 v8bf;
typedef __attribute__((ext_vector_type(8)))  float  v8f;
typedef __attribute__((address_space(3)))    bf16_t* lds_bf16_ptr;

union AFrag { v16bf v; v8bf h[2]; };

#define SEQ_ 256
#define B_   128
#define INP_ 256
#define H_   512
#define OUT_ 512
#define G4H_ 2048

__device__ __forceinline__ float sigmoidf_(float x) {
  return 1.0f / (1.0f + __expf(-x));
}

// ---------------------------------------------------------------------------
// Elementwise f32 -> bf16
__global__ void to_bf16_kernel(bf16_t* __restrict__ dst,
                               const float* __restrict__ src, int n) {
  int id = blockIdx.x * blockDim.x + threadIdx.x;
  if (id < n) dst[id] = (bf16_t)src[id];
}

// src: [K][N] f32 (row-major)  ->  dst: [N][K] bf16 (row-major transpose)
__global__ void transpose_to_bf16_kernel(bf16_t* __restrict__ dst,
                                         const float* __restrict__ src,
                                         int K, int N) {
  int id = blockIdx.x * blockDim.x + threadIdx.x;
  if (id >= K * N) return;
  int n = id / K;
  int k = id - n * K;
  dst[id] = (bf16_t)src[(size_t)k * N + n];
}

// Zero c-state (f32) and h double-buffer slot 0 (bf16); n = 2*B*H
__global__ void init_state_kernel(float* __restrict__ Cst,
                                  bf16_t* __restrict__ h0, int n) {
  int id = blockIdx.x * blockDim.x + threadIdx.x;
  if (id < n) { Cst[id] = 0.0f; h0[id] = (bf16_t)0.0f; }
}

// ---------------------------------------------------------------------------
// C[M][N] = A[M][K] @ Bt[N][K]^T + bias[N]   (bf16 in, f32 accumulate/out)
// Block tile = 128(M) x 64(N): 8 waves split M and SHARE the B panel, which
// is staged 256-K-columns at a time into LDS via async-to-LDS loads.
// Requires M % 128 == 0, N % 64 == 0, K % 256 == 0.
__global__ void __launch_bounds__(256)
gemm_bf16_nt(const bf16_t* __restrict__ A,
             const bf16_t* __restrict__ Bt,
             const float*  __restrict__ bias,
             float* __restrict__ C,
             int M, int N, int K) {
  __shared__ bf16_t ldsB[64 * 256];  // 32 KB B-panel chunk
  const int lane = threadIdx.x & 31;
  const int wave = threadIdx.x >> 5;
  const int half = lane >> 4;  // which 16-lane half
  const int r    = lane & 15;
  const int tilesN = N >> 6;
  const int bn = blockIdx.x % tilesN;
  const int bm = blockIdx.x / tilesN;
  const int m0 = bm * 128 + wave * 16;
  const int n0 = bn * 64;

  const bf16_t* arow = A + (size_t)(m0 + r) * K;
  const unsigned ldsBase =
      (unsigned)(unsigned long long)(lds_bf16_ptr)&ldsB[0];

  v8f acc[4] = {};
  for (int kc = 0; kc < K; kc += 256) {
    // ---- async-stage Bt[n0..n0+63][kc..kc+255] into LDS ----
#pragma unroll
    for (int it = 0; it < 8; ++it) {
      const int e   = it * 256 + (int)threadIdx.x;  // 16-byte chunk id
      const int row = e >> 5;                       // 512 B per row
      const int c16 = e & 31;
      const unsigned lofs = ldsBase + (unsigned)e * 16u;
      const bf16_t* g = Bt + (size_t)(n0 + row) * K + kc + c16 * 8;
      asm volatile("global_load_async_to_lds_b128 %0, %1, off"
                   :: "v"(lofs), "v"(g) : "memory");
    }
    asm volatile("s_wait_asynccnt 0x0" ::: "memory");
    __syncthreads();

    // ---- compute on the staged chunk ----
    for (int kb = 0; kb < 256; kb += 32) {
      AFrag a;
      a.h[0] = *(const v8bf*)(arow + kc + kb + half * 16);
      a.h[1] = *(const v8bf*)(arow + kc + kb + half * 16 + 8);
      __builtin_prefetch(arow + kc + kb + 64, 0, 0);
#pragma unroll
      for (int s = 0; s < 4; ++s) {
        const bf16_t* brow = &ldsB[(s * 16 + r) * 256 + kb + half * 16];
        v16bf b = *(const v16bf*)brow;
        acc[s] = __builtin_amdgcn_wmma_f32_16x16x32_bf16(
            false, a.v, false, b, (short)0, acc[s], false, false);
      }
    }
    __syncthreads();
  }

#pragma unroll
  for (int s = 0; s < 4; ++s) {
    const int col = n0 + s * 16 + r;
    const float bv = bias[col];
#pragma unroll
    for (int v = 0; v < 8; ++v) {
      C[(size_t)(m0 + half * 8 + v) * N + col] = acc[s][v] + bv;
    }
  }
}

// ---------------------------------------------------------------------------
// One LSTM timestep, both directions. 64 blocks x 8 waves:
//   block = (dir, unit-tile); waves = the 8 batch tiles, so all 8 waves read
//   the SAME Wh panel (one L2 fetch per panel per step). Each wave computes
//   the 4 gate tiles g = h@Wh for its 16x16 (batch,unit) tile, adds xp + bh,
//   applies the nonlinearity, updates c/h and records h into hs (bf16).
__global__ void __launch_bounds__(256)
lstm_step(const float*  __restrict__ xp,    // [2][SEQ][B][4H] (bx folded in)
          const bf16_t* __restrict__ whT,   // [2][4H][H]
          const float*  __restrict__ bh_f,  // [4H]
          const float*  __restrict__ bh_b,  // [4H]
          const bf16_t* __restrict__ h_in,  // [2][B][H]
          bf16_t* __restrict__ h_out,       // [2][B][H]
          float*  __restrict__ Hst,         // [2][B][H]
          float*  __restrict__ Cst,         // [2][B][H]
          bf16_t* __restrict__ hs,          // [SEQ][B][2H]
          int s) {
  const int lane = threadIdx.x & 31;
  const int wave = threadIdx.x >> 5;
  const int half = lane >> 4;
  const int r    = lane & 15;
  const int dir = blockIdx.x >> 5;  // 0 = fwd, 1 = bwd
  const int ut  = blockIdx.x & 31;  // hidden-unit tile
  const int m0  = wave << 4;        // batch tile (waves share the B operand)
  const int u0  = ut << 4;
  const int tt  = dir ? (SEQ_ - 1 - s) : s;

  const bf16_t* arow = h_in + ((size_t)dir * B_ + (m0 + r)) * H_;
  v8f acc[4] = {};
  for (int kb = 0; kb < H_; kb += 32) {
    AFrag a;
    a.h[0] = *(const v8bf*)(arow + kb + half * 16);
    a.h[1] = *(const v8bf*)(arow + kb + half * 16 + 8);
#pragma unroll
    for (int g = 0; g < 4; ++g) {
      const bf16_t* brow =
          whT + ((size_t)dir * G4H_ + g * H_ + u0 + r) * H_ + kb + half * 16;
      v16bf b = *(const v16bf*)brow;
      acc[g] = __builtin_amdgcn_wmma_f32_16x16x32_bf16(
          false, a.v, false, b, (short)0, acc[g], false, false);
    }
  }

  const float* bh = dir ? bh_b : bh_f;
  const int u = u0 + r;
#pragma unroll
  for (int v = 0; v < 8; ++v) {
    const int bi = m0 + half * 8 + v;
    const float* xr = xp + (((size_t)dir * SEQ_ + tt) * B_ + bi) * G4H_;
    const float gi = acc[0][v] + xr[u]          + bh[u];
    const float gf = acc[1][v] + xr[H_ + u]     + bh[H_ + u];
    const float go = acc[2][v] + xr[2 * H_ + u] + bh[2 * H_ + u];
    const float gc = acc[3][v] + xr[3 * H_ + u] + bh[3 * H_ + u];
    const size_t si = ((size_t)dir * B_ + bi) * H_ + u;
    const float cn = sigmoidf_(gf) * Cst[si] + sigmoidf_(gi) * tanhf(gc);
    const float hn = sigmoidf_(go) * tanhf(cn);
    Cst[si] = cn;
    Hst[si] = hn;
    h_out[si] = (bf16_t)hn;
    hs[((size_t)tt * B_ + bi) * (2 * H_) + dir * H_ + u] = (bf16_t)hn;
  }
}

// ---------------------------------------------------------------------------
extern "C" void kernel_launch(void* const* d_in, const int* in_sizes, int n_in,
                              void* d_out, int out_size, void* d_ws,
                              size_t ws_size, hipStream_t stream) {
  const float* x    = (const float*)d_in[0];
  const float* wx_f = (const float*)d_in[1];
  const float* bx_f = (const float*)d_in[2];
  const float* wh_f = (const float*)d_in[3];
  const float* bh_f = (const float*)d_in[4];
  const float* wx_b = (const float*)d_in[5];
  const float* bx_b = (const float*)d_in[6];
  const float* wh_b = (const float*)d_in[7];
  const float* bh_b = (const float*)d_in[8];
  const float* wfc  = (const float*)d_in[9];
  const float* bfc  = (const float*)d_in[10];
  float* out = (float*)d_out;

  // ---- workspace carve (256B aligned blocks) ----
  char* base = (char*)d_ws;
  size_t off = 0;
  auto carve = [&](size_t bytes) -> void* {
    void* q = base + off;
    off += (bytes + 255) & ~(size_t)255;
    return q;
  };
  float*  Hst  = (float*) carve((size_t)2 * B_ * H_ * 4);      // [2][B][H] f32
  float*  Cst  = (float*) carve((size_t)2 * B_ * H_ * 4);      // contiguous after Hst
  bf16_t* hbuf = (bf16_t*)carve((size_t)2 * 2 * B_ * H_ * 2);  // [2 buf][2 dir][B][H]
  bf16_t* xbf  = (bf16_t*)carve((size_t)SEQ_ * B_ * INP_ * 2);
  bf16_t* wxT  = (bf16_t*)carve((size_t)2 * G4H_ * INP_ * 2);  // [2][4H][INP]
  bf16_t* whT  = (bf16_t*)carve((size_t)2 * G4H_ * H_ * 2);    // [2][4H][H]
  bf16_t* wfcT = (bf16_t*)carve((size_t)OUT_ * 2 * H_ * 2);    // [OUT][2H]
  float*  xp   = (float*) carve((size_t)2 * SEQ_ * B_ * G4H_ * 4);
  bf16_t* hs   = (bf16_t*)carve((size_t)SEQ_ * B_ * 2 * H_ * 2);

  // ---- 1. conversions / transposes ----
  {
    int n = SEQ_ * B_ * INP_;
    to_bf16_kernel<<<(n + 255) / 256, 256, 0, stream>>>(xbf, x, n);
  }
  {
    int n = INP_ * G4H_;
    transpose_to_bf16_kernel<<<(n + 255) / 256, 256, 0, stream>>>(wxT, wx_f, INP_, G4H_);
    transpose_to_bf16_kernel<<<(n + 255) / 256, 256, 0, stream>>>(wxT + n, wx_b, INP_, G4H_);
  }
  {
    int n = H_ * G4H_;
    transpose_to_bf16_kernel<<<(n + 255) / 256, 256, 0, stream>>>(whT, wh_f, H_, G4H_);
    transpose_to_bf16_kernel<<<(n + 255) / 256, 256, 0, stream>>>(whT + n, wh_b, H_, G4H_);
  }
  {
    int n = 2 * H_ * OUT_;
    transpose_to_bf16_kernel<<<(n + 255) / 256, 256, 0, stream>>>(wfcT, wfc, 2 * H_, OUT_);
  }
  {
    int n = 2 * B_ * H_;
    init_state_kernel<<<(n + 255) / 256, 256, 0, stream>>>(Cst, hbuf, n);
  }

  // ---- 2. input projections (parallel WMMA GEMMs, LDS-staged B) ----
  {
    const int M = SEQ_ * B_, N = G4H_, K = INP_;
    const int blocks = (M / 128) * (N / 64);
    gemm_bf16_nt<<<blocks, 256, 0, stream>>>(xbf, wxT, bx_f, xp, M, N, K);
    gemm_bf16_nt<<<blocks, 256, 0, stream>>>(
        xbf, wxT + (size_t)G4H_ * INP_, bx_b, xp + (size_t)M * N, M, N, K);
  }

  // ---- 3. recurrence: one launch per timestep (both dirs fused) ----
  for (int s = 0; s < SEQ_; ++s) {
    bf16_t* hin  = hbuf + (size_t)(s & 1) * (2 * B_ * H_);
    bf16_t* hout = hbuf + (size_t)((s + 1) & 1) * (2 * B_ * H_);
    lstm_step<<<64, 256, 0, stream>>>(xp, whT, bh_f, bh_b, hin, hout,
                                      Hst, Cst, hs, s);
  }

  // ---- 4. final FC over merged hidden states ----
  {
    const int M = SEQ_ * B_, N = OUT_, K = 2 * H_;
    const int blocks = (M / 128) * (N / 64);
    gemm_bf16_nt<<<blocks, 256, 0, stream>>>(hs, wfcT, bfc, out, M, N, K);
  }

  // ---- 5. tail outputs: hN_f, hN_b, cN_f, cN_b (Hst then Cst, contiguous) ----
  hipMemcpyAsync(out + (size_t)SEQ_ * B_ * OUT_, Hst,
                 (size_t)4 * B_ * H_ * sizeof(float),
                 hipMemcpyDeviceToDevice, stream);
}
